// MultiHeadAttentionWithRoPE_43276090475189
// MI455X (gfx1250) — matrix operand order repaired and checked by reading, not compiled, not run on hardware
//
#include <hip/hip_runtime.h>
#include <hip/hip_bf16.h>
#include <math.h>

// ---------------- problem constants ----------------
constexpr int NB = 2;          // batch
constexpr int S  = 2048;       // sequence
constexpr int D  = 1024;       // model dim
constexpr int H  = 16;         // heads
constexpr int HD = 64;         // head dim
constexpr int M4 = NB * S;     // 4096 rows of flattened activations
#define LN_THETA 9.210340371976184f

typedef __attribute__((ext_vector_type(16))) __bf16 v16bf;
typedef __attribute__((ext_vector_type(8)))  float  v8f;

// ---------------- helpers ----------------
__device__ __forceinline__ unsigned short f2bf(float f) {
    union { float f; unsigned int u; } c; c.f = f;
    unsigned int u = c.u;
    unsigned int r = u + 0x7FFFu + ((u >> 16) & 1u);   // round-to-nearest-even
    return (unsigned short)(r >> 16);
}

// A-operand fragment (16x32 bf16, MxK): lane L holds row m=L&15; half h=L>>4.
// VGPRs 0..3 hold k = 8h..8h+7 ; VGPRs 4..7 hold k = 16+8h..16+8h+7.
__device__ __forceinline__ v16bf load_fragA(const unsigned short* rowbase, int h) {
    union { uint4 u[2]; v16bf v; } r;
    r.u[0] = *(const uint4*)(rowbase + 8 * h);
    r.u[1] = *(const uint4*)(rowbase + 16 + 8 * h);
    return r.v;
}

// B-operand fragment (32x16 bf16, KxN): lane L holds col n=L&15; half h=L>>4
// holds k = 8h..8h+7 and 16+8h..16+8h+7. rowbase = n-th row of B^T (k-major).
__device__ __forceinline__ v16bf load_fragB(const unsigned short* rowbase, int h) {
    union { uint4 u[2]; v16bf v; } r;
    r.u[0] = *(const uint4*)(rowbase + 8 * h);
    r.u[1] = *(const uint4*)(rowbase + 16 + 8 * h);
    return r.v;
}

__device__ __forceinline__ v8f wmma_bf16(v16bf a, v16bf b, v8f c) {
    return __builtin_amdgcn_wmma_f32_16x16x32_bf16(
        false, a, false, b, (short)0, c, false, false);
}

__device__ __forceinline__ float rowmax16(float x) {
    #pragma unroll
    for (int o = 1; o < 16; o <<= 1) x = fmaxf(x, __shfl_xor(x, o));
    return x;
}
__device__ __forceinline__ float rowsum16(float x) {
    #pragma unroll
    for (int o = 1; o < 16; o <<= 1) x += __shfl_xor(x, o);
    return x;
}

// Async memory->LDS staging (CDNA5 GLOBAL_LOAD_ASYNC_TO_LDS_B128, ASYNCcnt).
// Per ISA 10.2 aperture rules, the low 32 bits of a generic pointer into
// __shared__ are the wave-relative LDS byte offset.
__device__ __forceinline__ void async_copy_b128(unsigned short* ldsDst,
                                                const unsigned short* gSrc) {
    unsigned lds = (unsigned)(unsigned long long)ldsDst;
    unsigned long long ga = (unsigned long long)gSrc;
    asm volatile("global_load_async_to_lds_b128 %0, %1, off"
                 :: "v"(lds), "v"(ga) : "memory");
}
__device__ __forceinline__ void wait_async0() {
    asm volatile("s_wait_asynccnt 0" ::: "memory");
}
// Leave the next tile's 4 in-flight loads pending (in-order completion
// guarantees the previous tile's 4 loads are done).
__device__ __forceinline__ void wait_async4() {
    asm volatile("s_wait_asynccnt 4" ::: "memory");
}

// ---------------- phase 1: f32 -> bf16 convert ----------------
__global__ void cvt_bf16_kernel(const float* __restrict__ src,
                                unsigned short* __restrict__ dst, int n) {
    for (int i = blockIdx.x * blockDim.x + threadIdx.x; i < n;
         i += gridDim.x * blockDim.x) {
        dst[i] = f2bf(src[i]);
    }
}

// ---------------- phase 2 & 4: WMMA GEMM  C = A(4096x1024) @ W^T -------------
// MODE 0: blockIdx.z selects z=0 -> Qb (RoPE), z=1 -> Kb (RoPE),
//         z=2 -> Vt (store transposed per (b,h)).
// MODE 1: W = W0 (Wo), writes f32 to Of.
// Block: 256 threads (8 waves). Tile BM=128, BN=128, BK=32, double-buffered
// async LDS staging. Each wave computes 32x64 (2x4 accums, 8 wmma/K-step).
template <int MODE>
__global__ void __launch_bounds__(256)
gemm_bf16_kernel(const unsigned short* __restrict__ Abf,
                 const unsigned short* __restrict__ W0,
                 const unsigned short* __restrict__ W1,
                 const unsigned short* __restrict__ W2,
                 unsigned short* __restrict__ Qb,
                 unsigned short* __restrict__ Kb,
                 unsigned short* __restrict__ Vt,
                 float* __restrict__ Of,
                 const int* __restrict__ pos) {
    __shared__ unsigned short sA[2][128 * 32];   // 2 x 8 KB ping-pong
    __shared__ unsigned short sB[2][128 * 32];   // 2 x 8 KB ping-pong

    const int z = blockIdx.z;
    const unsigned short* W =
        (MODE == 1) ? W0 : (z == 0 ? W0 : (z == 1 ? W1 : W2));

    const int tM = blockIdx.x * 128;
    const int tN = blockIdx.y * 128;
    const int t    = threadIdx.x;
    const int lane = t & 31;
    const int wave = t >> 5;
    const int wm   = wave & 3;   // M offset 32*wm
    const int wn   = wave >> 2;  // N offset 64*wn
    const int lm   = lane & 15;
    const int lh   = lane >> 4;

    // this thread's two 16B staging chunks (row, col) within a 128x32 tile
    const int c0row = t >> 2;
    const int c0col = (t & 3) * 8;
    const int c1row = (t + 256) >> 2;
    const int c1col = c0col;

    auto stage = [&](int k0, int buf) {
        async_copy_b128(&sA[buf][c0row * 32 + c0col],
                        Abf + (size_t)(tM + c0row) * D + k0 + c0col);
        async_copy_b128(&sB[buf][c0row * 32 + c0col],
                        W + (size_t)(tN + c0row) * D + k0 + c0col);
        async_copy_b128(&sA[buf][c1row * 32 + c1col],
                        Abf + (size_t)(tM + c1row) * D + k0 + c1col);
        async_copy_b128(&sB[buf][c1row * 32 + c1col],
                        W + (size_t)(tN + c1row) * D + k0 + c1col);
    };

    v8f acc[2][4];
    #pragma unroll
    for (int i = 0; i < 2; ++i)
        #pragma unroll
        for (int j = 0; j < 4; ++j) acc[i][j] = (v8f){0,0,0,0,0,0,0,0};

    constexpr int NIT = D / 32;
    stage(0, 0);
    for (int it = 0; it < NIT; ++it) {
        const int cur = it & 1;
        if (it + 1 < NIT) {        // prefetch next tile into the other buffer
            stage((it + 1) * 32, cur ^ 1);
            wait_async4();         // current tile resident; next still in flight
        } else {
            wait_async0();
        }
        __syncthreads();

        const unsigned short* pA = sA[cur];
        const unsigned short* pB = sB[cur];
        v16bf a0  = load_fragA(&pA[(wm * 32 + lm) * 32], lh);
        v16bf a1  = load_fragA(&pA[(wm * 32 + 16 + lm) * 32], lh);
        v16bf bf0 = load_fragB(&pB[(wn * 64 + 0 * 16 + lm) * 32], lh);
        v16bf bf1 = load_fragB(&pB[(wn * 64 + 1 * 16 + lm) * 32], lh);
        v16bf bf2 = load_fragB(&pB[(wn * 64 + 2 * 16 + lm) * 32], lh);
        v16bf bf3 = load_fragB(&pB[(wn * 64 + 3 * 16 + lm) * 32], lh);

        acc[0][0] = wmma_bf16(a0, bf0, acc[0][0]);
        acc[0][1] = wmma_bf16(a0, bf1, acc[0][1]);
        acc[0][2] = wmma_bf16(a0, bf2, acc[0][2]);
        acc[0][3] = wmma_bf16(a0, bf3, acc[0][3]);
        acc[1][0] = wmma_bf16(a1, bf0, acc[1][0]);
        acc[1][1] = wmma_bf16(a1, bf1, acc[1][1]);
        acc[1][2] = wmma_bf16(a1, bf2, acc[1][2]);
        acc[1][3] = wmma_bf16(a1, bf3, acc[1][3]);
        __syncthreads();   // protects buffer `cur` from being restaged at it+2
    }

    // ---- epilogue: C layout => lane holds col n=lm, VGPR e holds row e+8*lh
    #pragma unroll
    for (int mi = 0; mi < 2; ++mi) {
        #pragma unroll
        for (int ni = 0; ni < 4; ++ni) {
            #pragma unroll
            for (int e = 0; e < 8; ++e) {
                int gm = tM + wm * 32 + mi * 16 + e + 8 * lh;
                int gn = tN + wn * 64 + ni * 16 + lm;
                float v = acc[mi][ni][e];
                if (MODE == 1) {
                    Of[(size_t)gm * D + gn] = v;
                } else if (z < 2) {
                    // fused RoPE (adjacent columns live in adjacent lanes)
                    int   sIdx = gm & (S - 1);
                    int   p    = pos[sIdx];
                    int   d    = gn & (HD - 1);
                    int   i2   = d >> 1;
                    float invf = __expf(-(float)(2 * i2) * (LN_THETA / (float)HD));
                    float ang  = (float)p * invf;
                    float sn, cs;
                    __sincosf(ang, &sn, &cs);
                    float prt = __shfl_xor(v, 1);
                    float r   = (d & 1) ? (prt * sn + v * cs) : (v * cs - prt * sn);
                    unsigned short* dst = (z == 0) ? Qb : Kb;
                    dst[(size_t)gm * D + gn] = f2bf(r);
                } else {
                    // V: store transposed per (b,h):  Vt[(bh*64 + d) * S + s]
                    int b    = gm >> 11;            // /S
                    int sIdx = gm & (S - 1);
                    int hh   = gn >> 6;
                    int d    = gn & (HD - 1);
                    Vt[((size_t)(b * H + hh) * HD + d) * S + sIdx] = f2bf(v);
                }
            }
        }
    }
}

// ---------------- phase 3: causal flash attention ----------------
// grid: (S/64, NB*H); block: 128 (4 waves). Each wave owns 16 query rows.
__global__ void __launch_bounds__(128)
flash_attn_kernel(const unsigned short* __restrict__ Qb,
                  const unsigned short* __restrict__ Kb,
                  const unsigned short* __restrict__ Vt,
                  unsigned short* __restrict__ Ob) {
    __shared__ unsigned short sP[4][16 * 32];   // per-wave P transpose patch

    const int bh   = blockIdx.y;
    const int b    = bh >> 4;       // /H
    const int hh   = bh & (H - 1);
    const int wave = threadIdx.x >> 5;
    const int lane = threadIdx.x & 31;
    const int lm   = lane & 15;
    const int lh   = lane >> 4;
    const int qs   = blockIdx.x * 64 + wave * 16;

    // Q fragments for this wave's 16 rows (d = 0..31 and 32..63)
    const unsigned short* qrow = Qb + ((size_t)(b * S + qs + lm)) * D + hh * HD;
    v16bf qa0 = load_fragA(qrow, lh);
    v16bf qa1 = load_fragA(qrow + 32, lh);

    v8f acc[4];
    #pragma unroll
    for (int i = 0; i < 4; ++i) acc[i] = (v8f){0,0,0,0,0,0,0,0};
    float mx[8], ls[8];
    #pragma unroll
    for (int e = 0; e < 8; ++e) { mx[e] = -3.0e38f; ls[e] = 0.f; }

    unsigned short* pb = &sP[wave][0];

    for (int j0 = 0; j0 <= qs + 15; j0 += 32) {
        // ---- scores S = Q K^T for a 32-key tile (two 16-col sub-tiles)
        const unsigned short* krow0 = Kb + ((size_t)(b * S + j0 + lm)) * D + hh * HD;
        const unsigned short* krow1 = krow0 + (size_t)16 * D;
        v16bf kb00 = load_fragB(krow0, lh);
        v16bf kb01 = load_fragB(krow0 + 32, lh);
        v16bf kb10 = load_fragB(krow1, lh);
        v16bf kb11 = load_fragB(krow1 + 32, lh);

        v8f s0 = (v8f){0,0,0,0,0,0,0,0};
        v8f s1 = (v8f){0,0,0,0,0,0,0,0};
        s0 = wmma_bf16(qa0, kb00, s0);
        s0 = wmma_bf16(qa1, kb01, s0);
        s1 = wmma_bf16(qa0, kb10, s1);
        s1 = wmma_bf16(qa1, kb11, s1);

        // ---- scale + causal mask + online softmax
        const int ja = j0 + lm;
        const int jb = j0 + 16 + lm;
        float nm[8];
        #pragma unroll
        for (int e = 0; e < 8; ++e) {
            int qr = qs + e + 8 * lh;
            float a = (ja <= qr) ? s0[e] * 0.125f : -3.0e38f;
            float c = (jb <= qr) ? s1[e] * 0.125f : -3.0e38f;
            s0[e] = a; s1[e] = c;
            float rm = rowmax16(fmaxf(a, c));
            nm[e] = fmaxf(mx[e], rm);
        }
        #pragma unroll
        for (int e = 0; e < 8; ++e) {
            float p0 = __expf(s0[e] - nm[e]);
            float p1 = __expf(s1[e] - nm[e]);
            s0[e] = p0; s1[e] = p1;
            float sc = __expf(mx[e] - nm[e]);
            float rs = rowsum16(p0 + p1);
            ls[e] = ls[e] * sc + rs;
            mx[e] = nm[e];
            #pragma unroll
            for (int dc = 0; dc < 4; ++dc) acc[dc][e] *= sc;
        }

        // ---- P: C-layout -> A-layout via per-wave LDS patch
        #pragma unroll
        for (int e = 0; e < 8; ++e) {
            int m = e + 8 * lh;
            pb[m * 32 + lm]      = f2bf(s0[e]);
            pb[m * 32 + 16 + lm] = f2bf(s1[e]);
        }
        asm volatile("s_wait_dscnt 0" ::: "memory");   // wave-local LDS RAW
        v16bf pa = load_fragA(pb + lm * 32, lh);

        // ---- O += P @ V  (V transposed in memory: Vt[d][s])
        #pragma unroll
        for (int dc = 0; dc < 4; ++dc) {
            const unsigned short* vrow =
                Vt + ((size_t)(bh * HD + dc * 16 + lm)) * S + j0;
            v16bf vb = load_fragB(vrow, lh);
            acc[dc] = wmma_bf16(pa, vb, acc[dc]);
        }
    }

    // ---- normalize and store (standard [4096][1024] bf16 layout)
    #pragma unroll
    for (int dc = 0; dc < 4; ++dc) {
        #pragma unroll
        for (int e = 0; e < 8; ++e) {
            float o = acc[dc][e] / ls[e];
            int gr = b * S + qs + e + 8 * lh;
            int gc = hh * HD + dc * 16 + lm;
            Ob[(size_t)gr * D + gc] = f2bf(o);
        }
    }
}

// ---------------- launch ----------------
extern "C" void kernel_launch(void* const* d_in, const int* in_sizes, int n_in,
                              void* d_out, int out_size, void* d_ws, size_t ws_size,
                              hipStream_t stream) {
    const float* Xf  = (const float*)d_in[0];
    const int*   pos = (const int*)d_in[1];
    const float* Wq  = (const float*)d_in[2];
    const float* Wk  = (const float*)d_in[3];
    const float* Wv  = (const float*)d_in[4];
    const float* Wo  = (const float*)d_in[5];
    float*       Of  = (float*)d_out;

    char* ws = (char*)d_ws;
    const size_t MB = 1024 * 1024;
    unsigned short* Xb  = (unsigned short*)(ws +  0 * MB);  // 8 MB
    unsigned short* Wqb = (unsigned short*)(ws +  8 * MB);  // 2 MB
    unsigned short* Wkb = (unsigned short*)(ws + 10 * MB);
    unsigned short* Wvb = (unsigned short*)(ws + 12 * MB);
    unsigned short* Wob = (unsigned short*)(ws + 14 * MB);
    unsigned short* Qbf = (unsigned short*)(ws + 16 * MB);  // 8 MB
    unsigned short* Kbf = (unsigned short*)(ws + 24 * MB);  // 8 MB
    unsigned short* Vt  = (unsigned short*)(ws + 32 * MB);  // 8 MB (transposed)
    unsigned short* Obf = (unsigned short*)(ws + 40 * MB);  // 8 MB

    // phase 1: convert everything to bf16 once
    cvt_bf16_kernel<<<512, 256, 0, stream>>>(Xf, Xb, M4 * D);
    cvt_bf16_kernel<<<256, 256, 0, stream>>>(Wq, Wqb, D * D);
    cvt_bf16_kernel<<<256, 256, 0, stream>>>(Wk, Wkb, D * D);
    cvt_bf16_kernel<<<256, 256, 0, stream>>>(Wv, Wvb, D * D);
    cvt_bf16_kernel<<<256, 256, 0, stream>>>(Wo, Wob, D * D);

    // phase 2: Q/K/V projections (+RoPE, V transposed)
    gemm_bf16_kernel<0><<<dim3(M4 / 128, D / 128, 3), 256, 0, stream>>>(
        Xb, Wqb, Wkb, Wvb, Qbf, Kbf, Vt, nullptr, pos);

    // phase 3: causal flash attention
    flash_attn_kernel<<<dim3(S / 64, NB * H), 128, 0, stream>>>(Qbf, Kbf, Vt, Obf);

    // phase 4: output projection -> f32
    gemm_bf16_kernel<1><<<dim3(M4 / 128, D / 128, 1), 256, 0, stream>>>(
        Obf, Wob, nullptr, nullptr, nullptr, nullptr, nullptr, Of, pos);
}
